// GRNGOB_53163105189928
// MI455X (gfx1250) — compile-verified, matrix operand order checked by reading, not run on hardware
//
#include <hip/hip_runtime.h>
#include <math.h>

// Problem constants (from reference)
#define ND 400
#define NS 300
#define NT 300
#define DH 256   // == SH == TH
#define ED 64
#define GH 512
#define GC 320   // DH+ED (gate output width)
#define XD_DRU 1664   // DH+DH+GC+GC+GH
#define XD_ST  1344   // DH+DH+GC+GH

// Padded dims so every GEMM has M,N % 32 == 0 and K % 64 == 0.
#define NDP 448      // 400 -> 448 (7*64)
#define NSP 320      // 300 -> 320 (5*64)
#define NTP 320
#define EPM 32       // edge-emb rows padded 6 -> 32

typedef __attribute__((ext_vector_type(16))) _Float16 v16h;
typedef __attribute__((ext_vector_type(8)))  float    v8f;

__device__ __forceinline__ float sigm_(float x) { return 1.0f / (1.0f + expf(-x)); }

// ---------------------------------------------------------------------------
// Tiled WMMA GEMM (no bounds checks): C[M,N] = A[M,K](lda) @ B[K,N](ldb) + bias
// Caller guarantees M,N multiples of 32 and K multiple of 64; all pointers and
// leading dims give 16B-aligned float4 accesses.
// 128 threads = 4 waves (2x2) per 32x32 output tile; k-step 64 => 2 WMMA per
// barrier pair. A staged [m][k], B staged transposed [n][k] in LDS as f16 so
// every fragment read is a contiguous 16B run (ds_load_b128).
// ---------------------------------------------------------------------------
#define KSTEP 64
#define LPAD 8       // LDS row stride = 72 halfs = 144B (16B aligned)
__global__ __launch_bounds__(128)
void gemm_wmma_f16(const float* __restrict__ A, int lda,
                   const float* __restrict__ B, int ldb,
                   float* __restrict__ C, int ldc,
                   int K, const float* __restrict__ bias)
{
    __shared__ _Float16 Als[32][KSTEP + LPAD];   // [m][k]
    __shared__ _Float16 Bls[32][KSTEP + LPAD];   // [n][k] (transposed)

    const int tid  = threadIdx.x;
    const int lane = tid & 31;
    const int wid  = tid >> 5;
    const int m0 = blockIdx.y * 32;
    const int n0 = blockIdx.x * 32;

    const int ar  = tid >> 2;            // 0..31  A row in tile
    const int ac  = (tid & 3) << 4;      // 0,16,32,48  A k-chunk (16 floats)
    const int bkr = tid >> 1;            // 0..63  B k-row in tile
    const int bnc = (tid & 1) << 4;      // 0,16   B n-chunk (16 floats)

    const int wrow = (wid >> 1) << 4;    // 0 or 16
    const int wcol = (wid & 1) << 4;     // 0 or 16
    const int fr   = lane & 15;
    const int kg   = (lane >> 4) << 3;   // 0 or 8

    const float* Ap = A + (size_t)(m0 + ar) * lda + ac;
    const float* Bp = B + (size_t)bkr * ldb + n0 + bnc;

    v8f acc = {0.f, 0.f, 0.f, 0.f, 0.f, 0.f, 0.f, 0.f};

    for (int k0 = 0; k0 < K; k0 += KSTEP) {
        // prefetch next k-block (global_prefetch_b8)
        if (k0 + KSTEP < K) {
            __builtin_prefetch(Ap + k0 + KSTEP, 0, 1);
            __builtin_prefetch(Bp + (size_t)(k0 + KSTEP) * ldb, 0, 1);
        }
        // ---- stage A tile: 4x float4, contiguous f16 stores ----
        {
            const float4* p = (const float4*)(Ap + k0);
#pragma unroll
            for (int q = 0; q < 4; ++q) {
                float4 x = p[q];
                Als[ar][ac + q * 4 + 0] = (_Float16)x.x;
                Als[ar][ac + q * 4 + 1] = (_Float16)x.y;
                Als[ar][ac + q * 4 + 2] = (_Float16)x.z;
                Als[ar][ac + q * 4 + 3] = (_Float16)x.w;
            }
        }
        // ---- stage B tile transposed ----
        {
            const float4* p = (const float4*)(Bp + (size_t)k0 * ldb);
#pragma unroll
            for (int q = 0; q < 4; ++q) {
                float4 x = p[q];
                Bls[bnc + q * 4 + 0][bkr] = (_Float16)x.x;
                Bls[bnc + q * 4 + 1][bkr] = (_Float16)x.y;
                Bls[bnc + q * 4 + 2][bkr] = (_Float16)x.z;
                Bls[bnc + q * 4 + 3][bkr] = (_Float16)x.w;
            }
        }
        __syncthreads();

        // ---- 2 WMMAs over the 64-wide k block ----
#pragma unroll
        for (int kk = 0; kk < KSTEP; kk += 32) {
            v16h a, b;
#pragma unroll
            for (int e = 0; e < 8; ++e) {
                a[e]     = Als[wrow + fr][kk + kg + e];
                a[e + 8] = Als[wrow + fr][kk + kg + 16 + e];
                b[e]     = Bls[wcol + fr][kk + kg + e];
                b[e + 8] = Bls[wcol + fr][kk + kg + 16 + e];
            }
            acc = __builtin_amdgcn_wmma_f32_16x16x32_f16(
                      false, a, false, b, (short)0, acc, false, false);
        }
        __syncthreads();
    }

    const int col = n0 + wcol + fr;
    const float bb = bias ? bias[col] : 0.0f;
#pragma unroll
    for (int v = 0; v < 8; ++v) {
        int row = m0 + wrow + kg + v;
        C[(size_t)row * ldc + col] = acc[v] + bb;
    }
}

// ---------------------------------------------------------------------------
// Elementwise / utility kernels
// ---------------------------------------------------------------------------
// Padded mask build: out is rp x cp, real data rows x cols, pads = 0.
__global__ void build_mask_pad_k(const int* __restrict__ m, float* __restrict__ out,
                                 int rows, int cols, int rp, int cp, int eq)
{
    int i = blockIdx.x * blockDim.x + threadIdx.x;
    if (i >= rp * cp) return;
    int r = i / cp, c = i - r * cp;
    float v = 0.0f;
    if (r < rows && c < cols) v = (m[(size_t)r * cols + c] == eq) ? 1.0f : 0.0f;
    out[i] = v;
}

// Copy edgeemb (6 x ED) into zero-padded EPM x ED buffer.
__global__ void pad_edge_k(const float* __restrict__ e, float* __restrict__ out)
{
    int i = blockIdx.x * blockDim.x + threadIdx.x;
    if (i >= EPM * ED) return;
    out[i] = (i < 6 * ED) ? e[i] : 0.0f;
}

__global__ void zero_k(float* __restrict__ p, int n)
{
    int i = blockIdx.x * blockDim.x + threadIdx.x;
    if (i < n) p[i] = 0.0f;
}

// dst[r, coff + c] = src[r, c]
__global__ void copy_cols_k(float* __restrict__ dst, int dld, int coff,
                            const float* __restrict__ src, int sld,
                            int rows, int cols)
{
    int i = blockIdx.x * blockDim.x + threadIdx.x;
    if (i >= rows * cols) return;
    int r = i / cols, c = i - r * cols;
    dst[(size_t)r * dld + coff + c] = src[(size_t)r * sld + c];
}

// dst[r, coff + c] = vec[c]
__global__ void bcast_cols_k(float* __restrict__ dst, int dld, int coff,
                             const float* __restrict__ vec, int rows, int cols)
{
    int i = blockIdx.x * blockDim.x + threadIdx.x;
    if (i >= rows * cols) return;
    int r = i / cols, c = i - r * cols;
    dst[(size_t)r * dld + coff + c] = vec[c];
}

// Decomposed gated message pass:
// out[o,c] = sum_j [type!=0] * he(j,type,c) * sigmoid(Aout[o,c]+Bin[j,c]+E[type,c])
__global__ void gated_agg_k(float* __restrict__ out,
                            const float* __restrict__ Aout,
                            const float* __restrict__ Bin,
                            const float* __restrict__ E,
                            const float* __restrict__ h_in,
                            const float* __restrict__ edgeemb,
                            const int* __restrict__ typemat,
                            int ld, int trans, int n_outer, int n_inner)
{
    __shared__ int s_type[448];          // n_inner <= 400
    int o = blockIdx.x;
    for (int j = threadIdx.x; j < n_inner; j += blockDim.x)
        s_type[j] = trans ? typemat[(size_t)j * ld + o] : typemat[(size_t)o * ld + j];
    __syncthreads();

    int c = threadIdx.x;                 // 0..GC-1
    if (o >= n_outer || c >= GC) return;
    float a = Aout[(size_t)o * GC + c];
    float s = 0.0f;
    for (int j = 0; j < n_inner; ++j) {
        int t = s_type[j];
        if (t != 0) {
            float logit = a + Bin[(size_t)j * GC + c] + E[(size_t)t * GC + c];
            float g = sigm_(logit);
            float he = (c < DH) ? h_in[(size_t)j * DH + c]
                                : edgeemb[(size_t)t * ED + (c - DH)];
            s += he * g;
        }
    }
    out[(size_t)o * GC + c] = s;
}

// h_new = z*h + (1-z)*tanh(cx + r*ux), r=sigmoid(rx), z=sigmoid(zx)
__global__ void gru_combine_k(const float* __restrict__ rx, const float* __restrict__ zx,
                              const float* __restrict__ cx, const float* __restrict__ ux,
                              const float* __restrict__ h, float* __restrict__ h_new, int n)
{
    int i = blockIdx.x * blockDim.x + threadIdx.x;
    if (i >= n) return;
    float r = sigm_(rx[i]);
    float z = sigm_(zx[i]);
    float u = tanhf(cx[i] + r * ux[i]);
    h_new[i] = z * h[i] + (1.0f - z) * u;
}

__global__ void mean_rows_k(const float* __restrict__ X, int rows, int cols,
                            float* __restrict__ out)
{
    int c = blockIdx.x * blockDim.x + threadIdx.x;
    if (c >= cols) return;
    float s = 0.0f;
    for (int r = 0; r < rows; ++r) s += X[(size_t)r * cols + c];
    out[c] = s / (float)rows;
}

// Global-state GRU: x(768), g(512). One block of 512 threads.
__global__ void g_gru_k(const float* __restrict__ x, const float* __restrict__ g,
                        const float* __restrict__ wr, const float* __restrict__ br,
                        const float* __restrict__ wz, const float* __restrict__ bz,
                        const float* __restrict__ wc, const float* __restrict__ bc,
                        const float* __restrict__ wu, const float* __restrict__ bu,
                        float* __restrict__ g_new)
{
    int o = threadIdx.x;
    if (o >= GH) return;
    float r = br[o], z = bz[o], c = bc[o], u = bu[o];
    for (int k = 0; k < 3 * DH; ++k) {
        float xv = x[k];
        r += xv * wr[(size_t)k * GH + o];
        z += xv * wz[(size_t)k * GH + o];
        c += xv * wc[(size_t)k * GH + o];
    }
    for (int k = 0; k < GH; ++k) {
        float gv = g[k];
        r += gv * wr[(size_t)(3 * DH + k) * GH + o];
        z += gv * wz[(size_t)(3 * DH + k) * GH + o];
        u += gv * wu[(size_t)k * GH + o];
    }
    r = sigm_(r);
    z = sigm_(z);
    float uu = tanhf(c + r * u);
    g_new[o] = z * g[o] + (1.0f - z) * uu;
}

__global__ void write_out_k(const float* __restrict__ dru_h, const float* __restrict__ g,
                            float* __restrict__ out)
{
    int i = blockIdx.x * blockDim.x + threadIdx.x;
    const int nd = ND * DH;
    if (i < nd) out[i] = dru_h[i];
    else if (i < nd + GH) out[i] = g[i - nd];
}

// ---------------------------------------------------------------------------
// Host side
// ---------------------------------------------------------------------------
static inline void gemm(const float* A, int lda, const float* B, int ldb,
                        float* C, int ldc, int M, int N, int K,
                        const float* bias, hipStream_t st)
{
    dim3 grid(N / 32, M / 32);           // M,N multiples of 32 by construction
    gemm_wmma_f16<<<grid, 128, 0, st>>>(A, lda, B, ldb, C, ldc, K, bias);
}

static inline int gsz(int n) { return (n + 255) / 256; }

// Input index map. Positional arrays in setup_inputs() insertion order, then
// `params` leaves in jax tree_flatten order (nested dict keys sorted).
enum {
    IN_DRU_EMB = 0, IN_DIS_EMB, IN_TAR_EMB,
    IN_M_DD, IN_M_SS, IN_M_TT, IN_M_DS, IN_M_DT,
    P_DISG_C_B, P_DISG_C_W, P_DISG_R_B, P_DISG_R_W,
    P_DISG_U_B, P_DISG_U_W, P_DISG_Z_B, P_DISG_Z_W,          // 8..15
    P_DRUG_C_B, P_DRUG_C_W, P_DRUG_R_B, P_DRUG_R_W,
    P_DRUG_U_B, P_DRUG_U_W, P_DRUG_Z_B, P_DRUG_Z_W,          // 16..23
    P_EDGEEMB,                                               // 24
    P_GG_C_B, P_GG_C_W, P_GG_R_B, P_GG_R_W,
    P_GG_U_B, P_GG_U_W, P_GG_Z_B, P_GG_Z_W,                  // 25..32
    P_G1_B, P_G1_W, P_G2_B, P_G2_W, P_G3_B, P_G3_W, P_G4_B, P_G4_W,   // 33..40
    P_TARG_C_B, P_TARG_C_W, P_TARG_R_B, P_TARG_R_W,
    P_TARG_U_B, P_TARG_U_W, P_TARG_Z_B, P_TARG_Z_W           // 41..48
};

extern "C" void kernel_launch(void* const* d_in, const int* in_sizes, int n_in,
                              void* d_out, int out_size, void* d_ws, size_t ws_size,
                              hipStream_t stream)
{
    (void)in_sizes; (void)n_in; (void)out_size; (void)ws_size;

    const float* dru_emb = (const float*)d_in[IN_DRU_EMB];
    const float* dis_emb = (const float*)d_in[IN_DIS_EMB];
    const float* tar_emb = (const float*)d_in[IN_TAR_EMB];
    const int*   m_dd = (const int*)d_in[IN_M_DD];
    const int*   m_ss = (const int*)d_in[IN_M_SS];
    const int*   m_tt = (const int*)d_in[IN_M_TT];
    const int*   m_ds = (const int*)d_in[IN_M_DS];
    const int*   m_dt = (const int*)d_in[IN_M_DT];
    const float* edgeemb = (const float*)d_in[P_EDGEEMB];
    const float* W1 = (const float*)d_in[P_G1_W]; const float* b1 = (const float*)d_in[P_G1_B];
    const float* W2 = (const float*)d_in[P_G2_W]; const float* b2 = (const float*)d_in[P_G2_B];
    const float* W3 = (const float*)d_in[P_G3_W]; const float* b3 = (const float*)d_in[P_G3_B];
    const float* W4 = (const float*)d_in[P_G4_W]; const float* b4 = (const float*)d_in[P_G4_B];

    // Workspace carve-out (all fp32; every block is a multiple of 4 floats,
    // preserving 16B alignment for the float4 GEMM paths).
    float* ws = (float*)d_ws;
    size_t off = 0;
    auto alloc = [&](size_t n) { float* p = ws + off; off += n; return p; };

    float* mask_dd = alloc((size_t)NDP * NDP);
    float* mask_ss = alloc((size_t)NSP * NSP);
    float* mask_tt = alloc((size_t)NTP * NTP);
    float* edge_p  = alloc((size_t)EPM * ED);
    float* dru_hA = alloc((size_t)NDP * DH); float* dru_hB = alloc((size_t)NDP * DH);
    float* dis_hA = alloc((size_t)NSP * DH); float* dis_hB = alloc((size_t)NSP * DH);
    float* tar_hA = alloc((size_t)NTP * DH); float* tar_hB = alloc((size_t)NTP * DH);
    float* gA = alloc(GH); float* gB = alloc(GH);
    float* dru_nn = alloc((size_t)NDP * DH);
    float* dis_nn = alloc((size_t)NSP * DH);
    float* tar_nn = alloc((size_t)NTP * DH);
    float* A2 = alloc((size_t)NDP * GC); float* B2 = alloc((size_t)NSP * GC); float* E2 = alloc((size_t)EPM * GC);
    float* A1 = alloc((size_t)NSP * GC); float* B1 = alloc((size_t)NDP * GC); float* E1 = alloc((size_t)EPM * GC);
    float* A3 = alloc((size_t)NTP * GC); float* B3 = alloc((size_t)NDP * GC); float* E3 = alloc((size_t)EPM * GC);
    float* A4 = alloc((size_t)NDP * GC); float* B4 = alloc((size_t)NTP * GC); float* E4 = alloc((size_t)EPM * GC);
    float* msg_dru_dis = alloc((size_t)ND * GC);
    float* msg_dis_dru = alloc((size_t)NS * GC);
    float* msg_tar_dru = alloc((size_t)NT * GC);
    float* msg_dru_tar = alloc((size_t)ND * GC);
    float* xcat_dru = alloc((size_t)NDP * (XD_DRU + DH));   // 448 x 1920
    float* xcat_dis = alloc((size_t)NSP * (XD_ST + DH));    // 320 x 1600
    float* xcat_tar = alloc((size_t)NTP * (XD_ST + DH));    // 320 x 1600
    float* rx = alloc((size_t)NDP * DH);
    float* zx = alloc((size_t)NDP * DH);
    float* cx = alloc((size_t)NDP * DH);
    float* ux = alloc((size_t)NDP * DH);
    float* meanv = alloc(3 * DH);

    // Padded masks (pads zero) and padded edge embedding
    build_mask_pad_k<<<gsz(NDP * NDP), 256, 0, stream>>>(m_dd, mask_dd, ND, ND, NDP, NDP, 1);
    build_mask_pad_k<<<gsz(NSP * NSP), 256, 0, stream>>>(m_ss, mask_ss, NS, NS, NSP, NSP, 2);
    build_mask_pad_k<<<gsz(NTP * NTP), 256, 0, stream>>>(m_tt, mask_tt, NT, NT, NTP, NTP, 3);
    pad_edge_k<<<gsz(EPM * ED), 256, 0, stream>>>(edgeemb, edge_p);

    // Zero initial hidden states (full padded extent -> pad rows stay zero)
    zero_k<<<gsz(NDP * DH), 256, 0, stream>>>(dru_hA, NDP * DH);
    zero_k<<<gsz(NSP * DH), 256, 0, stream>>>(dis_hA, NSP * DH);
    zero_k<<<gsz(NTP * DH), 256, 0, stream>>>(tar_hA, NTP * DH);
    zero_k<<<gsz(GH), 256, 0, stream>>>(gA, GH);
    // Zero pad rows of xcat buffers so GEMMs never read stale data
    zero_k<<<gsz((NDP - ND) * (XD_DRU + DH)), 256, 0, stream>>>(
        xcat_dru + (size_t)ND * (XD_DRU + DH), (NDP - ND) * (XD_DRU + DH));
    zero_k<<<gsz((NSP - NS) * (XD_ST + DH)), 256, 0, stream>>>(
        xcat_dis + (size_t)NS * (XD_ST + DH), (NSP - NS) * (XD_ST + DH));
    zero_k<<<gsz((NTP - NT) * (XD_ST + DH)), 256, 0, stream>>>(
        xcat_tar + (size_t)NT * (XD_ST + DH), (NTP - NT) * (XD_ST + DH));

    auto node_gru = [&](const float* xcat, int xw, int xd, const float* h,
                        int rows_p, int rows_real,
                        const float* Wr, const float* br_, const float* Wz, const float* bz_,
                        const float* Wc, const float* bc_, const float* Wu, const float* bu_,
                        float* h_new) {
        gemm(xcat, xw, Wr, DH, rx, DH, rows_p, DH, xw, br_, stream);   // [x,h] @ Wr
        gemm(xcat, xw, Wz, DH, zx, DH, rows_p, DH, xw, bz_, stream);   // [x,h] @ Wz
        gemm(xcat, xw, Wc, DH, cx, DH, rows_p, DH, xd, bc_, stream);   // x @ Wc
        gemm(h,    DH, Wu, DH, ux, DH, rows_p, DH, DH, bu_, stream);   // h @ Wu
        int n = rows_real * DH;
        gru_combine_k<<<gsz(n), 256, 0, stream>>>(rx, zx, cx, ux, h, h_new, n);
    };

    auto layer = [&](const float* dh, const float* sh, const float* th, const float* gv,
                     float* dh_n, float* sh_n, float* th_n, float* gv_n) {
        // 1. same-type sum aggregation (masked adjacency GEMMs, padded dims)
        gemm(mask_dd, NDP, dh, DH, dru_nn, DH, NDP, DH, NDP, nullptr, stream);
        gemm(mask_ss, NSP, sh, DH, dis_nn, DH, NSP, DH, NSP, nullptr, stream);
        gemm(mask_tt, NTP, th, DH, tar_nn, DH, NTP, DH, NTP, nullptr, stream);

        // 2. Gate projections (linear-over-concat decomposed into 3 GEMMs each)
        // gate2 W rows: [0,256)=dis_h, [256,320)=edge, [320,576)=dru_h ; outer=dru
        gemm(dh, DH, W2 + 320 * GC, GC, A2, GC, NDP, GC, DH, b2, stream);
        gemm(sh, DH, W2,            GC, B2, GC, NSP, GC, DH, nullptr, stream);
        gemm(edge_p, ED, W2 + 256 * GC, GC, E2, GC, EPM, GC, ED, nullptr, stream);
        // gate1 W rows: [0,256)=dru_h, [256,320)=edge, [320,576)=dis_h ; outer=dis
        gemm(sh, DH, W1 + 320 * GC, GC, A1, GC, NSP, GC, DH, b1, stream);
        gemm(dh, DH, W1,            GC, B1, GC, NDP, GC, DH, nullptr, stream);
        gemm(edge_p, ED, W1 + 256 * GC, GC, E1, GC, EPM, GC, ED, nullptr, stream);
        // gate3 W rows: [0,256)=dru_h, [256,320)=edge, [320,576)=tar_h ; outer=tar
        gemm(th, DH, W3 + 320 * GC, GC, A3, GC, NTP, GC, DH, b3, stream);
        gemm(dh, DH, W3,            GC, B3, GC, NDP, GC, DH, nullptr, stream);
        gemm(edge_p, ED, W3 + 256 * GC, GC, E3, GC, EPM, GC, ED, nullptr, stream);
        // gate4 W rows: [0,256)=tar_h, [256,320)=edge, [320,576)=dru_h ; outer=dru
        gemm(dh, DH, W4 + 320 * GC, GC, A4, GC, NDP, GC, DH, b4, stream);
        gemm(th, DH, W4,            GC, B4, GC, NTP, GC, DH, nullptr, stream);
        gemm(edge_p, ED, W4 + 256 * GC, GC, E4, GC, EPM, GC, ED, nullptr, stream);

        // 3. Gated masked aggregation (N^2 x 320 elementwise reduction)
        gated_agg_k<<<ND, GC, 0, stream>>>(msg_dru_dis, A2, B2, E2, sh, edgeemb, m_ds, NS, 0, ND, NS);
        gated_agg_k<<<NS, GC, 0, stream>>>(msg_dis_dru, A1, B1, E1, dh, edgeemb, m_ds, NS, 1, NS, ND);
        gated_agg_k<<<NT, GC, 0, stream>>>(msg_tar_dru, A3, B3, E3, dh, edgeemb, m_dt, NT, 1, NT, ND);
        gated_agg_k<<<ND, GC, 0, stream>>>(msg_dru_tar, A4, B4, E4, th, edgeemb, m_dt, NT, 0, ND, NT);

        // 4. Build GRU concat inputs (x parts ..., then h at the tail)
        const int WD = XD_DRU + DH;   // 1920
        copy_cols_k<<<gsz(ND * DH), 256, 0, stream>>>(xcat_dru, WD, 0,    dru_emb, DH, ND, DH);
        copy_cols_k<<<gsz(ND * DH), 256, 0, stream>>>(xcat_dru, WD, 256,  dru_nn,  DH, ND, DH);
        copy_cols_k<<<gsz(ND * GC), 256, 0, stream>>>(xcat_dru, WD, 512,  msg_dru_dis, GC, ND, GC);
        copy_cols_k<<<gsz(ND * GC), 256, 0, stream>>>(xcat_dru, WD, 832,  msg_dru_tar, GC, ND, GC);
        bcast_cols_k<<<gsz(ND * GH), 256, 0, stream>>>(xcat_dru, WD, 1152, gv, ND, GH);
        copy_cols_k<<<gsz(ND * DH), 256, 0, stream>>>(xcat_dru, WD, 1664, dh, DH, ND, DH);

        const int WS_ = XD_ST + DH;   // 1600
        copy_cols_k<<<gsz(NS * DH), 256, 0, stream>>>(xcat_dis, WS_, 0,    dis_emb, DH, NS, DH);
        copy_cols_k<<<gsz(NS * DH), 256, 0, stream>>>(xcat_dis, WS_, 256,  dis_nn,  DH, NS, DH);
        copy_cols_k<<<gsz(NS * GC), 256, 0, stream>>>(xcat_dis, WS_, 512,  msg_dis_dru, GC, NS, GC);
        bcast_cols_k<<<gsz(NS * GH), 256, 0, stream>>>(xcat_dis, WS_, 832, gv, NS, GH);
        copy_cols_k<<<gsz(NS * DH), 256, 0, stream>>>(xcat_dis, WS_, 1344, sh, DH, NS, DH);

        copy_cols_k<<<gsz(NT * DH), 256, 0, stream>>>(xcat_tar, WS_, 0,    tar_emb, DH, NT, DH);
        copy_cols_k<<<gsz(NT * DH), 256, 0, stream>>>(xcat_tar, WS_, 256,  tar_nn,  DH, NT, DH);
        copy_cols_k<<<gsz(NT * GC), 256, 0, stream>>>(xcat_tar, WS_, 512,  msg_tar_dru, GC, NT, GC);
        bcast_cols_k<<<gsz(NT * GH), 256, 0, stream>>>(xcat_tar, WS_, 832, gv, NT, GH);
        copy_cols_k<<<gsz(NT * DH), 256, 0, stream>>>(xcat_tar, WS_, 1344, th, DH, NT, DH);

        // 5. Node GRUs (WMMA GEMMs + elementwise combine)
        node_gru(xcat_dru, WD, XD_DRU, dh, NDP, ND,
                 (const float*)d_in[P_DRUG_R_W], (const float*)d_in[P_DRUG_R_B],
                 (const float*)d_in[P_DRUG_Z_W], (const float*)d_in[P_DRUG_Z_B],
                 (const float*)d_in[P_DRUG_C_W], (const float*)d_in[P_DRUG_C_B],
                 (const float*)d_in[P_DRUG_U_W], (const float*)d_in[P_DRUG_U_B], dh_n);
        node_gru(xcat_dis, WS_, XD_ST, sh, NSP, NS,
                 (const float*)d_in[P_DISG_R_W], (const float*)d_in[P_DISG_R_B],
                 (const float*)d_in[P_DISG_Z_W], (const float*)d_in[P_DISG_Z_B],
                 (const float*)d_in[P_DISG_C_W], (const float*)d_in[P_DISG_C_B],
                 (const float*)d_in[P_DISG_U_W], (const float*)d_in[P_DISG_U_B], sh_n);
        node_gru(xcat_tar, WS_, XD_ST, th, NTP, NT,
                 (const float*)d_in[P_TARG_R_W], (const float*)d_in[P_TARG_R_B],
                 (const float*)d_in[P_TARG_Z_W], (const float*)d_in[P_TARG_Z_B],
                 (const float*)d_in[P_TARG_C_W], (const float*)d_in[P_TARG_C_B],
                 (const float*)d_in[P_TARG_U_W], (const float*)d_in[P_TARG_U_B], th_n);

        // 6. Column means of the NEW hidden states (real rows only)
        mean_rows_k<<<1, DH, 0, stream>>>(dh_n, ND, DH, meanv + 0);
        mean_rows_k<<<1, DH, 0, stream>>>(sh_n, NS, DH, meanv + DH);
        mean_rows_k<<<1, DH, 0, stream>>>(th_n, NT, DH, meanv + 2 * DH);

        // 7. Global-state GRU
        g_gru_k<<<1, GH, 0, stream>>>(meanv, gv,
            (const float*)d_in[P_GG_R_W], (const float*)d_in[P_GG_R_B],
            (const float*)d_in[P_GG_Z_W], (const float*)d_in[P_GG_Z_B],
            (const float*)d_in[P_GG_C_W], (const float*)d_in[P_GG_C_B],
            (const float*)d_in[P_GG_U_W], (const float*)d_in[P_GG_U_B], gv_n);
    };

    // LAYERS == 2, ping-pong A -> B -> A
    layer(dru_hA, dis_hA, tar_hA, gA, dru_hB, dis_hB, tar_hB, gB);
    layer(dru_hB, dis_hB, tar_hB, gB, dru_hA, dis_hA, tar_hA, gA);

    int ntot = ND * DH + GH;
    write_out_k<<<gsz(ntot), 256, 0, stream>>>(dru_hA, gA, (float*)d_out);
}
